// Encoder_multi_76991583748156
// MI455X (gfx1250) — matrix operand order repaired and checked by reading, not compiled, not run on hardware
//
#include <hip/hip_runtime.h>
#include <hip/hip_bf16.h>
#include <cstdint>

typedef __attribute__((ext_vector_type(16))) _Float16 v16h;
typedef __attribute__((ext_vector_type(8)))  _Float16 v8h;
typedef __attribute__((ext_vector_type(8)))  float    v8f;

static constexpr int Bsz = 32;          // batch
static constexpr int Tsz = 256;         // time
static constexpr int Msz = Bsz * Tsz;   // 8192 rows
static constexpr int Esz = 512;         // embed dim
static constexpr int Hsz = 512;         // hidden
static constexpr int G4  = 4 * Hsz;     // 2048 gate rows

// LDS B-tile staging: 6 tiles (i/g/o x 2 dirs), 16 cols, 32 K-halves padded to 40.
static constexpr int LROW  = 40;               // halves per (tile,col) row (pad 32->40)
static constexpr int LTILE = 16 * LROW;        // 640 halves per tile
static constexpr int LSTG  = 6 * LTILE;        // 3840 halves per stage

#define USE_ASYNC_LDS 1

__device__ __forceinline__ float sig_(float x)  { return 1.0f / (1.0f + __expf(-x)); }
__device__ __forceinline__ float tanh_(float x) { return 2.0f / (1.0f + __expf(-2.0f * x)) - 1.0f; }

__device__ __forceinline__ v16h ld16(const _Float16* p0, const _Float16* p1) {
    union { v16h v; v8h h[2]; } u;
    u.h[0] = *(const v8h*)p0;
    u.h[1] = *(const v8h*)p1;
    return u.v;
}

// 16-byte global -> LDS copy. Async path uses the CDNA5 async-to-LDS DMA
// (tracked by ASYNCcnt); fallback stages through VGPRs (LOADcnt + DScnt).
__device__ __forceinline__ void cp16_g2l(const _Float16* g, _Float16* l) {
#if USE_ASYNC_LDS
    asm volatile("global_load_async_to_lds_b128 %0, %1, off"
                 :: "v"((unsigned)(uintptr_t)l), "v"(g)
                 : "memory");
#else
    *(v8h*)l = *(const v8h*)g;
#endif
}

__device__ __forceinline__ void async_fence() {
#if USE_ASYNC_LDS
    asm volatile("s_wait_asynccnt 0" ::: "memory");
#endif
}

__global__ void cvt_f32_to_f16(const float* __restrict__ src, _Float16* __restrict__ dst, int n) {
    int i = blockIdx.x * blockDim.x + threadIdx.x;
    if (i < n) dst[i] = (_Float16)src[i];
}

__global__ void embed_gather_f16(const int* __restrict__ x, const float* __restrict__ emb,
                                 _Float16* __restrict__ e16) {
    int i = blockIdx.x * blockDim.x + threadIdx.x;   // over Msz*Esz
    int row = i >> 9;
    int k   = i & 511;
    int tok = x[row];
    e16[i] = (_Float16)emb[(size_t)tok * Esz + k];
}

// Cooperative fill of one K-stage of the six B tiles into LDS.
// 6 tiles * 16 cols * 4 parts = 384 sixteen-byte chunks, 256 threads.
template<int K>
__device__ __forceinline__ void fill_stage(const _Float16* __restrict__ W, int n0, int kk,
                                           _Float16* sbuf, int tid) {
    #pragma unroll
    for (int rep = 0; rep < 2; ++rep) {
        int c = tid + rep * 256;
        if (c < 384) {
            int g    = c >> 6;        // 0..5
            int rem  = c & 63;
            int col  = rem >> 2;      // 0..15
            int part = rem & 3;       // 16B sub-chunk of the 64B K-row segment
            int dir  = (g >= 3) ? 1 : 0;
            int sub  = g - dir * 3;                       // 0:i 1:g 2:o
            int grow = dir * G4 + ((sub == 0) ? 0 : Hsz * (sub + 1));  // 0,2H,3H
            const _Float16* gp = W + (size_t)(grow + n0 + col) * K + kk + part * 8;
            _Float16* lp = sbuf + (g * 16 + col) * LROW + part * 8;
            cp16_g2l(gp, lp);
        }
    }
}

// Fused zero-state LSTM layer (both directions per wave, dead f-gate skipped).
// Block: 8 waves, 128 M-rows x 16 N-cols; weights double-buffered in LDS.
template<int K>
__global__ __launch_bounds__(256)
void lstm_layer_wmma(const _Float16* __restrict__ A,     // (Msz, K) f16
                     const _Float16* __restrict__ W,     // (2, G4, K) f16
                     const float* __restrict__ b_ih,     // (2, G4)
                     const float* __restrict__ b_hh,     // (2, G4)
                     _Float16* __restrict__ Hout,        // (Msz, 1024) or nullptr
                     float* __restrict__ out,            // d_out base
                     int layer) {
    __shared__ __align__(16) _Float16 smem[2][LSTG];

    const int tid  = threadIdx.x;
    const int lane = tid & 31;
    const int wave = tid >> 5;
    const int ln   = lane & 15;
    const int hh   = lane >> 4;                 // K-half select (ISA 7.12.2)

    const int nt = blockIdx.x & 31;             // 32 N-tiles
    const int mb = blockIdx.x >> 5;             // 64 M-blocks of 128 rows
    const int n0 = nt * 16;
    const int m0 = mb * 128 + wave * 16;

    constexpr int NIT = K / 32;

    // A operand: lane L holds row m0+(L%16); lanes 0-15 K {kk..+7, kk+16..+23},
    // lanes 16-31 K {kk+8..+15, kk+24..+31}.
    const _Float16* Ap = A + (size_t)(m0 + ln) * K + 8 * hh;

    fill_stage<K>(W, n0, 0, smem[0], tid);
    v16h a_cur = ld16(Ap, Ap + 16);

    v8f ci0 = {}, cg0 = {}, co0 = {}, ci1 = {}, cg1 = {}, co1 = {};

    for (int it = 0; it < NIT; ++it) {
        const int kk = it * 32;
        async_fence();            // my stage fills have landed in LDS
        __syncthreads();          // everyone's fills done; prior stage reads done
        if (it + 1 < NIT)
            fill_stage<K>(W, n0, kk + 32, smem[(it + 1) & 1], tid);

        // Prefetch next A fragment so the global load overlaps this step's WMMAs.
        v16h a_next = a_cur;
        if (it + 1 < NIT)
            a_next = ld16(Ap + kk + 32, Ap + kk + 48);

        // B fragments from LDS: lane -> column ln, K {8*hh..+7, 16+8*hh..+7}.
        const _Float16* bp = smem[it & 1] + ln * LROW + 8 * hh;
        v16h b0 = ld16(bp + 0 * LTILE, bp + 0 * LTILE + 16);
        v16h b1 = ld16(bp + 1 * LTILE, bp + 1 * LTILE + 16);
        v16h b2 = ld16(bp + 2 * LTILE, bp + 2 * LTILE + 16);
        v16h b3 = ld16(bp + 3 * LTILE, bp + 3 * LTILE + 16);
        v16h b4 = ld16(bp + 4 * LTILE, bp + 4 * LTILE + 16);
        v16h b5 = ld16(bp + 5 * LTILE, bp + 5 * LTILE + 16);

        ci0 = __builtin_amdgcn_wmma_f32_16x16x32_f16(false, a_cur, false, b0, (short)0, ci0, false, false);
        cg0 = __builtin_amdgcn_wmma_f32_16x16x32_f16(false, a_cur, false, b1, (short)0, cg0, false, false);
        co0 = __builtin_amdgcn_wmma_f32_16x16x32_f16(false, a_cur, false, b2, (short)0, co0, false, false);
        ci1 = __builtin_amdgcn_wmma_f32_16x16x32_f16(false, a_cur, false, b3, (short)0, ci1, false, false);
        cg1 = __builtin_amdgcn_wmma_f32_16x16x32_f16(false, a_cur, false, b4, (short)0, cg1, false, false);
        co1 = __builtin_amdgcn_wmma_f32_16x16x32_f16(false, a_cur, false, b5, (short)0, co1, false, false);

        a_cur = a_next;
    }

    // Per-lane gate biases (column ncol); gate order i,f,g,o -> offsets 0, 2H, 3H.
    const int ncol = n0 + ln;
    float bia0 = b_ih[ncol]                  + b_hh[ncol];
    float big0 = b_ih[2 * Hsz + ncol]        + b_hh[2 * Hsz + ncol];
    float bio0 = b_ih[3 * Hsz + ncol]        + b_hh[3 * Hsz + ncol];
    float bia1 = b_ih[G4 + ncol]             + b_hh[G4 + ncol];
    float big1 = b_ih[G4 + 2 * Hsz + ncol]   + b_hh[G4 + 2 * Hsz + ncol];
    float bio1 = b_ih[G4 + 3 * Hsz + ncol]   + b_hh[G4 + 3 * Hsz + ncol];

    // C/D layout: VGPR j holds M = m0 + 8*hh + j for this lane's column ncol.
    const int mrow0 = m0 + 8 * hh;
    #pragma unroll
    for (int j = 0; j < 8; ++j) {
        int row  = mrow0 + j;
        float c0 = sig_(ci0[j] + bia0) * tanh_(cg0[j] + big0);
        float h0 = sig_(co0[j] + bio0) * tanh_(c0);
        float c1 = sig_(ci1[j] + bia1) * tanh_(cg1[j] + big1);
        float h1 = sig_(co1[j] + bio1) * tanh_(c1);
        if (Hout) {
            Hout[(size_t)row * 1024 + ncol]       = (_Float16)h0;   // fwd
            Hout[(size_t)row * 1024 + 512 + ncol] = (_Float16)h1;   // bwd
        }
        // encoder_out at d_out offset 65536: (b*T + t, layer, n)
        out[65536 + ((size_t)row * 2 + layer) * Hsz + ncol] = h0 + h1;
        if ((row & (Tsz - 1)) == Tsz - 1) {        // t == T-1, forward dir
            int b = row >> 8;
            out[layer * (Bsz * Hsz) + b * Hsz + ncol]                 = h0;  // h_last
            out[2 * Bsz * Hsz + layer * (Bsz * Hsz) + b * Hsz + ncol] = c0;  // c_last
        }
    }
}

extern "C" void kernel_launch(void* const* d_in, const int* in_sizes, int n_in,
                              void* d_out, int out_size, void* d_ws, size_t ws_size,
                              hipStream_t stream) {
    const int*   x    = (const int*)  d_in[0];
    const float* emb  = (const float*)d_in[1];
    const float* Wih0 = (const float*)d_in[2];
    // d_in[3] = W_hh_l0 : unused (h0 == 0)
    const float* bih0 = (const float*)d_in[4];
    const float* bhh0 = (const float*)d_in[5];
    const float* Wih1 = (const float*)d_in[6];
    // d_in[7] = W_hh_l1 : unused
    const float* bih1 = (const float*)d_in[8];
    const float* bhh1 = (const float*)d_in[9];
    float* out = (float*)d_out;

    char* ws = (char*)d_ws;
    _Float16* e16  = (_Float16*)(ws);                 //  8,388,608 B : (8192,512) f16
    _Float16* W016 = (_Float16*)(ws +  8388608);      //  4,194,304 B : (2,2048,512) f16
    _Float16* W116 = (_Float16*)(ws + 12582912);      //  8,388,608 B : (2,2048,1024) f16
    _Float16* inp1 = (_Float16*)(ws + 20971520);      // 16,777,216 B : (8192,1024) f16

    const int n0 = 2 * G4 * Esz;        // 2,097,152
    cvt_f32_to_f16<<<(n0 + 255) / 256, 256, 0, stream>>>(Wih0, W016, n0);
    const int n1 = 2 * G4 * 2 * Hsz;    // 4,194,304
    cvt_f32_to_f16<<<(n1 + 255) / 256, 256, 0, stream>>>(Wih1, W116, n1);

    embed_gather_f16<<<(Msz * Esz) / 256, 256, 0, stream>>>(x, emb, e16);

    // (Msz/128) M-blocks * (Hsz/16) N-tiles = 2048 blocks of 8 waves
    lstm_layer_wmma<512> <<<2048, 256, 0, stream>>>(e16,  W016, bih0, bhh0, inp1,    out, 0);
    lstm_layer_wmma<1024><<<2048, 256, 0, stream>>>(inp1, W116, bih1, bhh1, nullptr, out, 1);
}
